// EncodeBlock_9998683865179
// MI455X (gfx1250) — matrix operand order repaired and checked by reading, not compiled, not run on hardware
//
#include <hip/hip_runtime.h>
#include <hip/hip_bf16.h>

// ---------------------------------------------------------------------------
// Types for CDNA5 WMMA (wave32)
// ---------------------------------------------------------------------------
typedef __attribute__((ext_vector_type(16))) __bf16          v16bf;
typedef __attribute__((ext_vector_type(16))) unsigned short  u16v16;
typedef __attribute__((ext_vector_type(8)))  float           v8f;

// fp32 -> bf16 (round-to-nearest-even)
__device__ __forceinline__ unsigned short f2bf(float f) {
  unsigned int u = __builtin_bit_cast(unsigned int, f);
  u += 0x7FFFu + ((u >> 16) & 1u);
  return (unsigned short)(u >> 16);
}

// two fp32 -> packed bf16x2 in a u32 (lo = first, hi = second), RNE
__device__ __forceinline__ unsigned int f2bf2(float lo, float hi) {
  unsigned int a = __builtin_bit_cast(unsigned int, lo);
  unsigned int b = __builtin_bit_cast(unsigned int, hi);
  a += 0x7FFFu + ((a >> 16) & 1u);
  b += 0x7FFFu + ((b >> 16) & 1u);
  return (a >> 16) | (b & 0xFFFF0000u);
}

__device__ __forceinline__ v8f vzero8() {
  v8f z;
#pragma unroll
  for (int i = 0; i < 8; ++i) z[i] = 0.0f;
  return z;
}

// Async 16-byte global -> LDS copy (CDNA5, ASYNCcnt-tracked, no VGPR data).
// VDST VGPR holds the LDS byte address (low 32 bits of a generic LDS pointer).
__device__ __forceinline__ void async_copy16(const void* gptr, void* lptr) {
  const unsigned lds_addr = (unsigned)(size_t)lptr;
  asm volatile("global_load_async_to_lds_b128 %0, %1, off"
               :: "v"(lds_addr), "v"(gptr) : "memory");
}
__device__ __forceinline__ void wait_async0() {
  asm volatile("s_wait_asynccnt 0x0" ::: "memory");
}

// A-fragment (16x32, 16-bit): lane L -> row = base + (L&15); half = L>>4.
// elems 0..15 = K(16*half .. 16*half+15) contiguous
__device__ __forceinline__ u16v16 load_frag_a(const unsigned short* s, int ld,
                                              int baseRow, int colOff, int lane) {
  const int row  = baseRow + (lane & 15);
  const int half = lane >> 4;
  const unsigned short* p = s + row * ld + colOff;
  u16v16 a;
#pragma unroll
  for (int i = 0; i < 16; ++i) a[i] = p[16 * half + i];
  return a;
}

// B-fragment (32x16, 16-bit): lane L -> col = base + (L&15); half = L>>4.
// elems 0..15 = K(16*half .. 16*half+15) contiguous in the "col" row of s.
__device__ __forceinline__ u16v16 load_frag_b(const unsigned short* s, int ld,
                                              int baseCol, int colOff, int lane) {
  const int n    = baseCol + (lane & 15);
  const int half = lane >> 4;
  const unsigned short* p = s + n * ld + colOff + 16 * half;
  u16v16 b;
#pragma unroll
  for (int i = 0; i < 16; ++i) b[i] = p[i];
  return b;
}

__device__ __forceinline__ v8f wmma_bf16(u16v16 a, u16v16 b, v8f c) {
  return __builtin_amdgcn_wmma_f32_16x16x32_bf16(
      false, __builtin_bit_cast(v16bf, a),
      false, __builtin_bit_cast(v16bf, b),
      (short)0, c, false, false);
}

// ---------------------------------------------------------------------------
// fp32 -> bf16 bulk converter (one pass per tensor; bandwidth-trivial)
// ---------------------------------------------------------------------------
__global__ __launch_bounds__(256) void cvt_bf16_k(const float* __restrict__ in,
                                                  unsigned short* __restrict__ out,
                                                  int n) {
  const int i = (blockIdx.x * 256 + threadIdx.x) * 8;
  if (i >= n) return;
  const float4 a = *(const float4*)(in + i);
  const float4 b = *(const float4*)(in + i + 4);
  uint4 o;
  o.x = f2bf2(a.x, a.y);
  o.y = f2bf2(a.z, a.w);
  o.z = f2bf2(b.x, b.y);
  o.w = f2bf2(b.z, b.w);
  *(uint4*)(out + i) = o;
}

// ---------------------------------------------------------------------------
// GEMM:  C[M,N] = A[M,K] @ W[N,K]^T, A/W pre-converted bf16.
// EPI: 0 = write bf16 Cb, 1 = exact-erf GELU -> bf16 Cb, 2 = +res -> fp32 C
// WG: 256 threads (8 waves, 2Mx4N); tile 128x256; wave tile 64x64 = 16 WMMA.
// LDS fills use CDNA5 async global->LDS b128 copies, double buffered.
// ---------------------------------------------------------------------------
template <int EPI>
__global__ __launch_bounds__(256) void gemm_bf16_k(
    const unsigned short* __restrict__ A, const unsigned short* __restrict__ W,
    const float* __restrict__ res, float* __restrict__ C,
    unsigned short* __restrict__ Cb, int M, int N, int K) {
  __shared__ unsigned short As[2][128 * 32];
  __shared__ unsigned short Ws[2][256 * 32];

  const int tid  = threadIdx.x;
  const int lane = tid & 31;
  const int wid  = tid >> 5;
  const int wm   = wid & 1;   // 2 waves along M (64 rows each)
  const int wn   = wid >> 1;  // 4 waves along N (64 cols each)
  const int m0   = blockIdx.y * 128;
  const int n0   = blockIdx.x * 256;

  v8f acc[4][4];
#pragma unroll
  for (int i = 0; i < 4; ++i)
#pragma unroll
    for (int j = 0; j < 4; ++j) acc[i][j] = vzero8();

  // Async tile fill: A 128x32 (512 x 16B), W 256x32 (1024 x 16B)
  auto fill = [&](int k0, int buf) {
#pragma unroll
    for (int it = 0; it < 2; ++it) {
      const int chunk = it * 256 + tid;
      const int r = chunk >> 2, c8 = (chunk & 3) * 8;
      async_copy16(A + (size_t)(m0 + r) * K + k0 + c8, &As[buf][r * 32 + c8]);
    }
#pragma unroll
    for (int it = 0; it < 4; ++it) {
      const int chunk = it * 256 + tid;
      const int r = chunk >> 2, c8 = (chunk & 3) * 8;
      async_copy16(W + (size_t)(n0 + r) * K + k0 + c8, &Ws[buf][r * 32 + c8]);
    }
  };

  const int nsteps = K >> 5;
  fill(0, 0);
  for (int s = 0; s < nsteps; ++s) {
    wait_async0();      // this wave's fills for buf[s&1] have landed
    __syncthreads();    // every wave's fills have landed
    if (s + 1 < nsteps) fill((s + 1) << 5, (s + 1) & 1);

    const unsigned short* Ab = As[s & 1];
    const unsigned short* Wb = Ws[s & 1];
    u16v16 af[4], bfr[4];
#pragma unroll
    for (int i = 0; i < 4; ++i) af[i]  = load_frag_a(Ab, 32, wm * 64 + i * 16, 0, lane);
#pragma unroll
    for (int j = 0; j < 4; ++j) bfr[j] = load_frag_b(Wb, 32, wn * 64 + j * 16, 0, lane);
#pragma unroll
    for (int i = 0; i < 4; ++i)
#pragma unroll
      for (int j = 0; j < 4; ++j) acc[i][j] = wmma_bf16(af[i], bfr[j], acc[i][j]);
    __syncthreads();    // done reading buf[s&1] before it is refilled at s+2
  }

  const int ln = lane & 15, half = lane >> 4;
#pragma unroll
  for (int i = 0; i < 4; ++i) {
#pragma unroll
    for (int j = 0; j < 4; ++j) {
#pragma unroll
      for (int e = 0; e < 8; ++e) {
        const int row = m0 + wm * 64 + i * 16 + e + 8 * half;
        const int col = n0 + wn * 64 + j * 16 + ln;
        float v = acc[i][j][e];
        if (EPI == 1) v = 0.5f * v * (1.0f + erff(v * 0.70710678118654752f));
        if (EPI == 2) {
          C[(size_t)row * N + col] = v + res[(size_t)row * N + col];
        } else {
          Cb[(size_t)row * N + col] = f2bf(v);
        }
      }
    }
  }
}

// ---------------------------------------------------------------------------
// Flash attention: per (b,h); WG = 128 threads (4 waves), 64 query rows.
// q/k/v are bf16 in [B,S,DIM] layout (head h at cols h*64..h*64+63).
// Output attn written as bf16 (feeds the wo GEMM directly).
// ---------------------------------------------------------------------------
__global__ __launch_bounds__(128) void flash_attn_k(
    const unsigned short* __restrict__ q, const unsigned short* __restrict__ k,
    const unsigned short* __restrict__ v, const float* __restrict__ mask,
    unsigned short* __restrict__ attn) {
  const int S = 2048, DIMC = 1024, D = 64;
  __shared__ unsigned short Qs[64 * 64];
  __shared__ unsigned short Ks[32 * 64];
  __shared__ unsigned short Vt[64 * 32];  // transposed: [dim][key]
  __shared__ unsigned short Ps[4][16 * 32];

  const int tid  = threadIdx.x;
  const int lane = tid & 31;
  const int wid  = tid >> 5;
  const int q0   = blockIdx.x * 64;
  const int b    = blockIdx.y >> 4;
  const int h    = blockIdx.y & 15;

  const unsigned short* qb = q + (size_t)b * S * DIMC + h * D;
  const unsigned short* kp = k + (size_t)b * S * DIMC + h * D;
  const unsigned short* vp = v + (size_t)b * S * DIMC + h * D;

  // Q tile 64x64 bf16 via async copies (512 x 16B chunks, 4 per thread)
#pragma unroll
  for (int it = 0; it < 4; ++it) {
    const int chunk = it * 128 + tid;
    const int r = chunk >> 3, c8 = (chunk & 7) * 8;
    async_copy16(qb + (size_t)(q0 + r) * DIMC + c8, &Qs[r * 64 + c8]);
  }
  wait_async0();
  __syncthreads();
  const u16v16 qa0 = load_frag_a(Qs, 64, wid * 16, 0,  lane);
  const u16v16 qa1 = load_frag_a(Qs, 64, wid * 16, 32, lane);

  const int ln = lane & 15, half = lane >> 4;
  float m_[8], l_[8];
#pragma unroll
  for (int e = 0; e < 8; ++e) { m_[e] = -3.0e38f; l_[e] = 0.0f; }
  v8f o0 = vzero8(), o1 = vzero8(), o2 = vzero8(), o3 = vzero8();
  v8f* oacc[4] = {&o0, &o1, &o2, &o3};

  for (int kbk = 0; kbk < S / 32; ++kbk) {
    __syncthreads();
    // K block 32x64 bf16 via async; V block transposed via regular loads
#pragma unroll
    for (int it = 0; it < 2; ++it) {
      const int chunk = it * 128 + tid;
      const int r = chunk >> 3, c8 = (chunk & 7) * 8;
      async_copy16(kp + (size_t)(kbk * 32 + r) * DIMC + c8, &Ks[r * 64 + c8]);
      const uint4 g = *(const uint4*)(vp + (size_t)(kbk * 32 + r) * DIMC + c8);
      Vt[(c8 + 0) * 32 + r] = (unsigned short)(g.x & 0xFFFF);
      Vt[(c8 + 1) * 32 + r] = (unsigned short)(g.x >> 16);
      Vt[(c8 + 2) * 32 + r] = (unsigned short)(g.y & 0xFFFF);
      Vt[(c8 + 3) * 32 + r] = (unsigned short)(g.y >> 16);
      Vt[(c8 + 4) * 32 + r] = (unsigned short)(g.z & 0xFFFF);
      Vt[(c8 + 5) * 32 + r] = (unsigned short)(g.z >> 16);
      Vt[(c8 + 6) * 32 + r] = (unsigned short)(g.w & 0xFFFF);
      Vt[(c8 + 7) * 32 + r] = (unsigned short)(g.w >> 16);
    }
    if (kbk + 1 < S / 32) {  // prefetch next K block into cache (global_prefetch_b8)
      __builtin_prefetch(kp + (size_t)((kbk + 1) * 32 + (tid >> 2)) * DIMC, 0, 3);
    }
    wait_async0();
    __syncthreads();

    // Scores: two 16x16 frags (keys 0-15, 16-31), contraction d=64
    v8f s0 = vzero8(), s1 = vzero8();
    {
      u16v16 bk;
      bk = load_frag_b(Ks, 64, 0,  0,  lane); s0 = wmma_bf16(qa0, bk, s0);
      bk = load_frag_b(Ks, 64, 0,  32, lane); s0 = wmma_bf16(qa1, bk, s0);
      bk = load_frag_b(Ks, 64, 16, 0,  lane); s1 = wmma_bf16(qa0, bk, s1);
      bk = load_frag_b(Ks, 64, 16, 32, lane); s1 = wmma_bf16(qa1, bk, s1);
    }

    // Online softmax (fp32; rows live in element index e)
    const int key0 = kbk * 32 + ln;
    float p0[8], p1[8], alpha[8];
#pragma unroll
    for (int e = 0; e < 8; ++e) {
      const int row = q0 + wid * 16 + e + 8 * half;
      float v0 = s0[e] * 0.125f + mask[(size_t)row * S + key0];
      float v1 = s1[e] * 0.125f + mask[(size_t)row * S + key0 + 16];
      float mx = fmaxf(v0, v1);
#pragma unroll
      for (int off = 1; off < 16; off <<= 1)
        mx = fmaxf(mx, __shfl_xor(mx, off, 32));
      const float mnew = fmaxf(m_[e], mx);
      alpha[e] = __expf(m_[e] - mnew);
      p0[e] = __expf(v0 - mnew);
      p1[e] = __expf(v1 - mnew);
      float sm = p0[e] + p1[e];
#pragma unroll
      for (int off = 1; off < 16; off <<= 1)
        sm += __shfl_xor(sm, off, 32);
      l_[e] = l_[e] * alpha[e] + sm;
      m_[e] = mnew;
    }
#pragma unroll
    for (int f = 0; f < 4; ++f)
#pragma unroll
      for (int e = 0; e < 8; ++e) (*oacc[f])[e] *= alpha[e];

    // P (16x32) -> LDS bf16, then P@V via WMMA
    unsigned short* Pw = Ps[wid];
#pragma unroll
    for (int e = 0; e < 8; ++e) {
      const int pr = e + 8 * half;
      Pw[pr * 32 + ln]      = f2bf(p0[e]);
      Pw[pr * 32 + 16 + ln] = f2bf(p1[e]);
    }
    __syncthreads();

    const u16v16 pa = load_frag_a(Pw, 32, 0, 0, lane);
#pragma unroll
    for (int f = 0; f < 4; ++f) {
      const u16v16 bv = load_frag_b(Vt, 32, 16 * f, 0, lane);
      *oacc[f] = wmma_bf16(pa, bv, *oacc[f]);
    }
  }

  // Normalize and store attn as bf16 ([B,S,DIM], head cols h*64..)
  unsigned short* ap = attn + (size_t)b * S * DIMC + h * D;
#pragma unroll
  for (int f = 0; f < 4; ++f) {
#pragma unroll
    for (int e = 0; e < 8; ++e) {
      const int row = q0 + wid * 16 + e + 8 * half;
      ap[(size_t)row * DIMC + 16 * f + ln] = f2bf((*oacc[f])[e] / l_[e]);
    }
  }
}

// ---------------------------------------------------------------------------
// LayerNorm (torch-style: std with ddof=1, divide by (std + eps))
// One 256-thread block per row of 1024. Optional bf16 mirror output.
// ---------------------------------------------------------------------------
__global__ __launch_bounds__(256) void layernorm_k(
    const float* __restrict__ x, const float* __restrict__ gain,
    const float* __restrict__ bias, float* __restrict__ out,
    unsigned short* __restrict__ outb, int N) {
  __shared__ float red[16];
  const int row = blockIdx.x;
  const int tid = threadIdx.x;
  const float* xr = x + (size_t)row * N;
  const float4 vv = *(const float4*)(xr + tid * 4);
  float s  = vv.x + vv.y + vv.z + vv.w;
  float sq = vv.x * vv.x + vv.y * vv.y + vv.z * vv.z + vv.w * vv.w;
#pragma unroll
  for (int off = 1; off < 32; off <<= 1) {
    s  += __shfl_xor(s, off, 32);
    sq += __shfl_xor(sq, off, 32);
  }
  const int lane = tid & 31, wid = tid >> 5;
  if (lane == 0) { red[wid] = s; red[8 + wid] = sq; }
  __syncthreads();
  s = 0.0f; sq = 0.0f;
#pragma unroll
  for (int i = 0; i < 8; ++i) { s += red[i]; sq += red[8 + i]; }
  const float mean = s / (float)N;
  float var = (sq - (float)N * mean * mean) / (float)(N - 1);
  var = fmaxf(var, 0.0f);
  const float inv = 1.0f / (sqrtf(var) + 1e-6f);
  const float4 aa = *(const float4*)(gain + tid * 4);
  const float4 bb = *(const float4*)(bias + tid * 4);
  float4 o;
  o.x = aa.x * (vv.x - mean) * inv + bb.x;
  o.y = aa.y * (vv.y - mean) * inv + bb.y;
  o.z = aa.z * (vv.z - mean) * inv + bb.z;
  o.w = aa.w * (vv.w - mean) * inv + bb.w;
  *(float4*)(out + (size_t)row * N + tid * 4) = o;
  if (outb) {
    uint2 ob;
    ob.x = f2bf2(o.x, o.y);
    ob.y = f2bf2(o.z, o.w);
    *(uint2*)(outb + (size_t)row * N + tid * 4) = ob;
  }
}

// ---------------------------------------------------------------------------
// Launch
// ---------------------------------------------------------------------------
extern "C" void kernel_launch(void* const* d_in, const int* in_sizes, int n_in,
                              void* d_out, int out_size, void* d_ws, size_t ws_size,
                              hipStream_t stream) {
  const float* x    = (const float*)d_in[0];
  const float* mask = (const float*)d_in[1];
  const float* wq   = (const float*)d_in[2];
  const float* wk   = (const float*)d_in[3];
  const float* wv   = (const float*)d_in[4];
  const float* wo   = (const float*)d_in[5];
  const float* w1   = (const float*)d_in[6];
  const float* w2   = (const float*)d_in[7];
  const float* a1   = (const float*)d_in[8];
  const float* b1   = (const float*)d_in[9];
  const float* a2   = (const float*)d_in[10];
  const float* b2   = (const float*)d_in[11];
  float* out = (float*)d_out;

  const int Bn = 2, S = 2048, DIMC = 1024, FFNC = 4096;
  const int M = Bn * S;                       // 4096 rows
  const size_t T  = (size_t)M * DIMC;         // 4M elems
  const size_t TW = (size_t)DIMC * DIMC;      // 1M elems
  const size_t TF = (size_t)M * FFNC;         // 16M elems
  const size_t TWF = (size_t)FFNC * DIMC;     // 4M elems

  // bf16 workspace region
  unsigned short* wsb = (unsigned short*)d_ws;
  unsigned short* xb    = wsb;                 // [M,DIM]
  unsigned short* wqb   = xb + T;              // [DIM,DIM]
  unsigned short* wkb   = wqb + TW;
  unsigned short* wvb   = wkb + TW;
  unsigned short* wob   = wvb + TW;
  unsigned short* w1b   = wob + TW;            // [FFN,DIM]
  unsigned short* w2b   = w1b + TWF;           // [DIM,FFN]
  unsigned short* qb    = w2b + TWF;           // [M,DIM]
  unsigned short* kb    = qb + T;
  unsigned short* vb    = kb + T;
  unsigned short* attnb = vb + T;              // [M,DIM]
  unsigned short* axb   = attnb + T;           // [M,DIM] bf16 mirror of attn_x
  unsigned short* hb    = axb + T;             // [M,FFN]
  // fp32 workspace region (after all bf16 buffers)
  float* wsf   = (float*)(hb + TF);
  float* tmp   = wsf;                          // [M,DIM] pre-LN sums
  float* attnx = tmp + T;                      // [M,DIM] fp32 attn_x

  const dim3 blk(256);
  const dim3 gProj(DIMC / 256, M / 128);  // (4, 32)
  const dim3 gFfn1(FFNC / 256, M / 128);  // (16, 32)

  // One-time fp32 -> bf16 conversions (x + all weights)
  cvt_bf16_k<<<dim3((int)(T   / 2048)), blk, 0, stream>>>(x,  xb,  (int)T);
  cvt_bf16_k<<<dim3((int)(TW  / 2048)), blk, 0, stream>>>(wq, wqb, (int)TW);
  cvt_bf16_k<<<dim3((int)(TW  / 2048)), blk, 0, stream>>>(wk, wkb, (int)TW);
  cvt_bf16_k<<<dim3((int)(TW  / 2048)), blk, 0, stream>>>(wv, wvb, (int)TW);
  cvt_bf16_k<<<dim3((int)(TW  / 2048)), blk, 0, stream>>>(wo, wob, (int)TW);
  cvt_bf16_k<<<dim3((int)(TWF / 2048)), blk, 0, stream>>>(w1, w1b, (int)TWF);
  cvt_bf16_k<<<dim3((int)(TWF / 2048)), blk, 0, stream>>>(w2, w2b, (int)TWF);

  // QKV projections (bf16 in, bf16 out)
  gemm_bf16_k<0><<<gProj, blk, 0, stream>>>(xb, wqb, nullptr, nullptr, qb, M, DIMC, DIMC);
  gemm_bf16_k<0><<<gProj, blk, 0, stream>>>(xb, wkb, nullptr, nullptr, kb, M, DIMC, DIMC);
  gemm_bf16_k<0><<<gProj, blk, 0, stream>>>(xb, wvb, nullptr, nullptr, vb, M, DIMC, DIMC);

  // Flash attention
  flash_attn_k<<<dim3(S / 64, Bn * 16), dim3(128), 0, stream>>>(qb, kb, vb, mask, attnb);

  // Output projection + residual (fp32), then LayerNorm 1 (fp32 + bf16 mirror)
  gemm_bf16_k<2><<<gProj, blk, 0, stream>>>(attnb, wob, x, tmp, nullptr, M, DIMC, DIMC);
  layernorm_k<<<dim3(M), blk, 0, stream>>>(tmp, a1, b1, attnx, axb, DIMC);

  // FFN: GEMM1 + exact GELU -> bf16 h; GEMM2 + residual -> fp32; LayerNorm 2
  gemm_bf16_k<1><<<gFfn1, blk, 0, stream>>>(axb, w1b, nullptr, nullptr, hb, M, FFNC, DIMC);
  gemm_bf16_k<2><<<gProj, blk, 0, stream>>>(hb, w2b, attnx, tmp, nullptr, M, DIMC, FFNC);
  layernorm_k<<<dim3(M), blk, 0, stream>>>(tmp, a2, b2, out, nullptr, DIMC);
}
